// PAGTN_1692217114868
// MI455X (gfx1250) — matrix operand order repaired and checked by reading, not compiled, not run on hardware
//
#include <hip/hip_runtime.h>
#include <hip/hip_bf16.h>

// ---------------- problem constants (fixed by the reference) ----------------
#define NN    30000
#define EE    480000
#define BB    256
#define FN    94
#define FE    14
#define HIDD  64
#define GF    200
#define PREDD 128
#define DEPTH 5
#define SLOPE 0.2f

typedef _Float16 half_t;
typedef half_t  v16h  __attribute__((ext_vector_type(16)));
typedef float   v8f   __attribute__((ext_vector_type(8)));
typedef unsigned int u32x4 __attribute__((ext_vector_type(4)));

__device__ __forceinline__ float lrelu(float x) { return x > 0.0f ? x : SLOPE * x; }

__device__ __forceinline__ float atomicMaxF(float* addr, float v) {
    if (v >= 0.0f)
        return __int_as_float(atomicMax((int*)addr, __float_as_int(v)));
    return __uint_as_float(atomicMin((unsigned int*)addr, __float_as_uint(v)));
}

// ---------------------------------------------------------------------------
// Prep: repack 5x [64,64] f32 weights (row-major out x in) into WMMA B-operand
// layout (f16), plus pack the 5 biases contiguously.
// B element j of lane l for tile (w, ot, ks):  B[k][n] = W[n][k]
//   n = ot*16 + (l&15),  k = ks*32 + (l>=16 ? 8 : 0) + (j < 8 ? j : 16 + (j - 8))
// ---------------------------------------------------------------------------
__global__ __launch_bounds__(256) void k_prep(
    const float* w0, const float* w1, const float* w2, const float* w3, const float* w4,
    const float* b0, const float* b1, const float* b2, const float* b3, const float* b4,
    half_t* Bp, float* Bias)
{
    int t = blockIdx.x * 256 + threadIdx.x;
    const float* Wa[5] = {w0, w1, w2, w3, w4};
    const float* Ba[5] = {b0, b1, b2, b3, b4};
    if (t < 5 * 4096) {
        int w = t / 4096, r = t % 4096;
        int ot = r / 1024; r %= 1024;
        int ks = r / 512;  r %= 512;
        int lane = r / 16, j = r % 16;
        int n = ot * 16 + (lane & 15);
        int k = ks * 32 + ((lane >= 16) ? 8 : 0) + (j < 8 ? j : 16 + (j - 8));
        Bp[t] = (half_t)Wa[w][n * HIDD + k];
    } else if (t < 5 * 4096 + 5 * HIDD) {
        int i = t - 5 * 4096;
        Bias[i] = Ba[i / HIDD][i % HIDD];
    }
}

// ---------------------------------------------------------------------------
// x0 = lrelu(node_feats @ atom_inp_w^T + b); also seeds h16 for layer 0.
// ---------------------------------------------------------------------------
__global__ __launch_bounds__(256) void k_x0(
    const float* __restrict__ nf, const float* __restrict__ W, const float* __restrict__ b,
    float* __restrict__ x0, half_t* __restrict__ h16)
{
    int idx = blockIdx.x * 256 + threadIdx.x;
    if (idx >= NN * HIDD) return;
    int n = idx >> 6, o = idx & 63;
    float acc = b[o];
    const float* wrow = W + o * FN;
    const float* frow = nf + (size_t)n * FN;
    #pragma unroll 2
    for (int i = 0; i < FN; ++i) acc += frow[i] * wrow[i];
    float v = lrelu(acc);
    x0[idx] = v;
    h16[idx] = (half_t)v;
}

// ---------------------------------------------------------------------------
// Per-layer zero/init of agg / node max / node sum.
// ---------------------------------------------------------------------------
__global__ __launch_bounds__(256) void k_init_layer(float* agg, float* nmax, float* nsum)
{
    int idx = blockIdx.x * 256 + threadIdx.x;
    if (idx >= NN * HIDD) return;
    agg[idx] = 0.0f;
    if ((idx & 63) == 0) {
        int n = idx >> 6;
        nmax[n] = __uint_as_float(0xff800000u); // -inf
        nsum[n] = 0.0f;
    }
}

// ---------------------------------------------------------------------------
// k_proj5: 5 projections of h [N,64] via WMMA f16->f32.
// One wave handles a 16-row tile; A loaded once (ISA 16-bit A layout),
// 5 weights x 4 out-tiles x 2 K-steps = 40 v_wmma per tile.
// Output P[w] = h @ W_w^T + b_w, w in {attn_src, attn_dst, msg_src, msg_dst, wgt_n}
// ---------------------------------------------------------------------------
__global__ __launch_bounds__(256) void k_proj5(
    const half_t* __restrict__ h16, const half_t* __restrict__ Bp,
    const float* __restrict__ Bias, float* __restrict__ P)
{
    int wid  = (blockIdx.x * 256 + threadIdx.x) >> 5;  // wave-uniform
    int lane = threadIdx.x & 31;
    if (wid >= NN / 16) return;                         // whole wave exits together
    int row0 = wid * 16;
    int m = lane & 15, hi = lane >> 4;

    // A operand: lane m holds row row0+m.
    //   elements 0..7  = K off+0..off+7,  elements 8..15 = K off+16..off+23, off = hi?8:0
    const half_t* rp = h16 + (size_t)(row0 + m) * HIDD;
    int off = hi ? 8 : 0;
    union UA { u32x4 u[2]; v16h v; };
    UA a0, a1;
    a0.u[0] = *(const u32x4*)(rp + off);
    a0.u[1] = *(const u32x4*)(rp + off + 16);
    a1.u[0] = *(const u32x4*)(rp + 32 + off);
    a1.u[1] = *(const u32x4*)(rp + 32 + off + 16);

    #pragma unroll
    for (int w = 0; w < 5; ++w) {
        float* O = P + (size_t)w * NN * HIDD;
        #pragma unroll
        for (int ot = 0; ot < 4; ++ot) {
            const half_t* bp = Bp + ((w * 4 + ot) * 2) * 512 + lane * 16;
            v16h b0 = *(const v16h*)(bp);
            v16h b1 = *(const v16h*)(bp + 512);
            v8f c = {0.f, 0.f, 0.f, 0.f, 0.f, 0.f, 0.f, 0.f};
            c = __builtin_amdgcn_wmma_f32_16x16x32_f16(false, a0.v, false, b0,
                                                       (short)0, c, false, false);
            c = __builtin_amdgcn_wmma_f32_16x16x32_f16(false, a1.v, false, b1,
                                                       (short)0, c, false, false);
            int   col  = ot * 16 + m;
            float bias = Bias[w * HIDD + col];
            #pragma unroll
            for (int r = 0; r < 8; ++r) {
                int row = row0 + (hi ? 8 + r : r);
                O[(size_t)row * HIDD + col] = c[r] + bias;
            }
        }
    }
}

// ---------------------------------------------------------------------------
// Edge scores: wave-per-edge. score = dot(lrelu(AS[src]+AD[dst]+e_atn), dotw)+b
// e_atn recomputed inline from 14-dim edge features. atomicMax into nmax[dst].
// ---------------------------------------------------------------------------
__global__ __launch_bounds__(256) void k_score(
    const float* __restrict__ P, const float* __restrict__ ef,
    const int* __restrict__ src, const int* __restrict__ dst,
    const float* __restrict__ aew, const float* __restrict__ aeb,
    const float* __restrict__ dotw, const float* __restrict__ dotb,
    float* __restrict__ score, float* __restrict__ nmax)
{
    int e = (blockIdx.x * 256 + threadIdx.x) >> 5;
    if (e >= EE) return;
    int lane = threadIdx.x & 31;
    const float* AS = P;
    const float* AD = P + (size_t)NN * HIDD;
    int s = src[e], d = dst[e];
    const float* fe = ef + (size_t)e * FE;
    float acc = 0.0f;
    #pragma unroll
    for (int h = 0; h < 2; ++h) {
        int k = lane + 32 * h;
        float ea = aeb[k];
        #pragma unroll
        for (int i = 0; i < FE; ++i) ea += fe[i] * aew[k * FE + i];
        float t = lrelu(AS[(size_t)s * HIDD + k] + AD[(size_t)d * HIDD + k] + ea);
        acc += t * dotw[k];
    }
    #pragma unroll
    for (int o = 16; o > 0; o >>= 1) acc += __shfl_xor(acc, o, 32);
    if (lane == 0) {
        float sc = acc + dotb[0];
        score[e] = sc;
        atomicMaxF(nmax + d, sc);
    }
}

// ---------------------------------------------------------------------------
// Edge softmax numerator + denominator accumulation.
// ---------------------------------------------------------------------------
__global__ __launch_bounds__(256) void k_expsum(
    const float* __restrict__ score, const int* __restrict__ dst,
    const float* __restrict__ nmax, float* __restrict__ exb, float* __restrict__ nsum)
{
    int e = blockIdx.x * 256 + threadIdx.x;
    if (e >= EE) return;
    int d = dst[e];
    float ex = __expf(score[e] - nmax[d]);
    exb[e] = ex;
    atomicAdd(nsum + d, ex);
}

// ---------------------------------------------------------------------------
// Weighted messages: thread per (edge, feature).
// msg = lrelu(MS[src]+MD[dst]+e_msg);  agg[dst] += softmax_w * msg
// ---------------------------------------------------------------------------
__global__ __launch_bounds__(256) void k_message(
    const float* __restrict__ P, const float* __restrict__ ef,
    const int* __restrict__ src, const int* __restrict__ dst,
    const float* __restrict__ mew, const float* __restrict__ meb,
    const float* __restrict__ exb, const float* __restrict__ nsum,
    float* __restrict__ agg)
{
    int idx = blockIdx.x * 256 + threadIdx.x;
    if (idx >= EE * HIDD) return;
    int e = idx >> 6, k = idx & 63;
    const float* MS = P + (size_t)2 * NN * HIDD;
    const float* MD = P + (size_t)3 * NN * HIDD;
    int s = src[e], d = dst[e];
    float w = exb[e] / nsum[d];
    const float* fe = ef + (size_t)e * FE;
    float em = meb[k];
    #pragma unroll
    for (int i = 0; i < FE; ++i) em += fe[i] * mew[k * FE + i];
    float mv = lrelu(MS[(size_t)s * HIDD + k] + MD[(size_t)d * HIDD + k] + em);
    atomicAdd(&agg[(size_t)d * HIDD + k], w * mv);
}

// ---------------------------------------------------------------------------
// Node update: h = relu(lrelu(agg + WN) + x0); write f32 and f16 views.
// ---------------------------------------------------------------------------
__global__ __launch_bounds__(256) void k_update(
    const float* __restrict__ agg, const float* __restrict__ P,
    const float* __restrict__ x0, float* __restrict__ h32, half_t* __restrict__ h16)
{
    int idx = blockIdx.x * 256 + threadIdx.x;
    if (idx >= NN * HIDD) return;
    const float* WN = P + (size_t)4 * NN * HIDD;
    float v = lrelu(agg[idx] + WN[idx]) + x0[idx];
    v = v > 0.0f ? v : 0.0f;
    h32[idx] = v;
    h16[idx] = (half_t)v;
}

// ---------------------------------------------------------------------------
// node_h = lrelu(concat(node_feats, h) @ atom_out_w^T + b)  -> [N, 200]
// ---------------------------------------------------------------------------
__global__ __launch_bounds__(256) void k_nodeh(
    const float* __restrict__ nf, const float* __restrict__ h32,
    const float* __restrict__ W, const float* __restrict__ b, float* __restrict__ nh)
{
    int idx = blockIdx.x * 256 + threadIdx.x;
    if (idx >= NN * GF) return;
    int n = idx / GF, o = idx % GF;
    float acc = b[o];
    const float* wr = W + (size_t)o * (FN + HIDD);
    const float* fr = nf + (size_t)n * FN;
    const float* hr = h32 + (size_t)n * HIDD;
    #pragma unroll 2
    for (int i = 0; i < FN; ++i) acc += fr[i] * wr[i];
    #pragma unroll 4
    for (int j = 0; j < HIDD; ++j) acc += hr[j] * wr[FN + j];
    nh[idx] = lrelu(acc);
}

// ---------------------------------------------------------------------------
// Per-node sigmoid gate  w = sigmoid(node_h @ w_sum_w + b)
// ---------------------------------------------------------------------------
__global__ __launch_bounds__(256) void k_wsum(
    const float* __restrict__ nh, const float* __restrict__ W,
    const float* __restrict__ b, float* __restrict__ wn)
{
    int n = blockIdx.x * 256 + threadIdx.x;
    if (n >= NN) return;
    float acc = b[0];
    const float* r = nh + (size_t)n * GF;
    #pragma unroll 4
    for (int i = 0; i < GF; ++i) acc += r[i] * W[i];
    wn[n] = 1.0f / (1.0f + __expf(-acc));
}

__global__ __launch_bounds__(256) void k_init_hg(float* hgs, float* hgm)
{
    int i = blockIdx.x * 256 + threadIdx.x;
    if (i >= BB * GF) return;
    hgs[i] = 0.0f;
    hgm[i] = __uint_as_float(0xff800000u);
}

// ---------------------------------------------------------------------------
// Readout: weighted-sum + max over the nodes of each graph.
// ---------------------------------------------------------------------------
__global__ __launch_bounds__(256) void k_readout(
    const float* __restrict__ nh, const float* __restrict__ wn,
    const int* __restrict__ n2g, float* __restrict__ hgs, float* __restrict__ hgm)
{
    int idx = blockIdx.x * 256 + threadIdx.x;
    if (idx >= NN * GF) return;
    int n = idx / GF, j = idx % GF;
    int g = n2g[n];
    float v = nh[idx];
    atomicAdd(&hgs[(size_t)g * GF + j], v * wn[n]);
    atomicMaxF(&hgm[(size_t)g * GF + j], v);
}

// ---------------------------------------------------------------------------
// Output transform: [B,400] @ transform_w^T + b  -> [B,128]
// ---------------------------------------------------------------------------
__global__ __launch_bounds__(256) void k_final(
    const float* __restrict__ hgs, const float* __restrict__ hgm,
    const float* __restrict__ W, const float* __restrict__ b, float* __restrict__ out)
{
    int idx = blockIdx.x * 256 + threadIdx.x;
    if (idx >= BB * PREDD) return;
    int bb = idx / PREDD, o = idx % PREDD;
    float acc = b[o];
    const float* wr = W + (size_t)o * (2 * GF);
    const float* sr = hgs + (size_t)bb * GF;
    const float* mr = hgm + (size_t)bb * GF;
    #pragma unroll 4
    for (int i = 0; i < GF; ++i) acc += sr[i] * wr[i];
    #pragma unroll 4
    for (int i = 0; i < GF; ++i) acc += mr[i] * wr[GF + i];
    out[idx] = acc;
}

// ---------------------------------------------------------------------------
extern "C" void kernel_launch(void* const* d_in, const int* in_sizes, int n_in,
                              void* d_out, int out_size, void* d_ws, size_t ws_size,
                              hipStream_t stream)
{
    const float* node_feats = (const float*)d_in[0];
    const float* edge_feats = (const float*)d_in[1];
    const int*   src        = (const int*)d_in[2];
    const int*   dst        = (const int*)d_in[3];
    const int*   n2g        = (const int*)d_in[4];
    // d_in[5] = num_graphs (compile-time BB)
    const float* atom_inp_w = (const float*)d_in[6];
    const float* atom_inp_b = (const float*)d_in[7];
    const float* attn_src_w = (const float*)d_in[8];
    const float* attn_src_b = (const float*)d_in[9];
    const float* attn_dst_w = (const float*)d_in[10];
    const float* attn_dst_b = (const float*)d_in[11];
    const float* attn_edg_w = (const float*)d_in[12];
    const float* attn_edg_b = (const float*)d_in[13];
    const float* attn_dot_w = (const float*)d_in[14];
    const float* attn_dot_b = (const float*)d_in[15];
    const float* msg_src_w  = (const float*)d_in[16];
    const float* msg_src_b  = (const float*)d_in[17];
    const float* msg_dst_w  = (const float*)d_in[18];
    const float* msg_dst_b  = (const float*)d_in[19];
    const float* msg_edg_w  = (const float*)d_in[20];
    const float* msg_edg_b  = (const float*)d_in[21];
    const float* wgt_n_w    = (const float*)d_in[22];
    const float* wgt_n_b    = (const float*)d_in[23];
    const float* atom_out_w = (const float*)d_in[24];
    const float* atom_out_b = (const float*)d_in[25];
    const float* w_sum_w    = (const float*)d_in[26];
    const float* w_sum_b    = (const float*)d_in[27];
    const float* trans_w    = (const float*)d_in[28];
    const float* trans_b    = (const float*)d_in[29];

    // ---- workspace carve-out (256B aligned slices) ----
    char* base = (char*)d_ws;
    size_t off = 0;
    auto carve = [&](size_t bytes) -> char* {
        char* p = base + off;
        off = (off + bytes + 255) & ~(size_t)255;
        return p;
    };
    float*  x0    = (float*)carve((size_t)NN * HIDD * 4);
    float*  h32   = (float*)carve((size_t)NN * HIDD * 4);
    half_t* h16   = (half_t*)carve((size_t)NN * HIDD * 2);
    float*  P     = (float*)carve((size_t)5 * NN * HIDD * 4);
    float*  agg   = (float*)carve((size_t)NN * HIDD * 4);
    float*  score = (float*)carve((size_t)EE * 4);
    float*  exb   = (float*)carve((size_t)EE * 4);
    float*  nmax  = (float*)carve((size_t)NN * 4);
    float*  nsum  = (float*)carve((size_t)NN * 4);
    float*  nh    = (float*)carve((size_t)NN * GF * 4);
    float*  wn    = (float*)carve((size_t)NN * 4);
    float*  hgs   = (float*)carve((size_t)BB * GF * 4);
    float*  hgm   = (float*)carve((size_t)BB * GF * 4);
    half_t* Bp    = (half_t*)carve((size_t)5 * 4096 * 2);
    float*  Bias  = (float*)carve((size_t)5 * HIDD * 4);

    auto cdiv = [](long a, long b) -> int { return (int)((a + b - 1) / b); };

    // weight repack for WMMA (once per call)
    k_prep<<<cdiv(5 * 4096 + 5 * HIDD, 256), 256, 0, stream>>>(
        attn_src_w, attn_dst_w, msg_src_w, msg_dst_w, wgt_n_w,
        attn_src_b, attn_dst_b, msg_src_b, msg_dst_b, wgt_n_b, Bp, Bias);

    // input projection -> x0, h16
    k_x0<<<cdiv((long)NN * HIDD, 256), 256, 0, stream>>>(
        node_feats, atom_inp_w, atom_inp_b, x0, h16);

    for (int layer = 0; layer < DEPTH; ++layer) {
        k_init_layer<<<cdiv((long)NN * HIDD, 256), 256, 0, stream>>>(agg, nmax, nsum);
        k_proj5<<<cdiv((long)(NN / 16), 8), 256, 0, stream>>>(h16, Bp, Bias, P);
        k_score<<<EE / 8, 256, 0, stream>>>(
            P, edge_feats, src, dst, attn_edg_w, attn_edg_b,
            attn_dot_w, attn_dot_b, score, nmax);
        k_expsum<<<cdiv(EE, 256), 256, 0, stream>>>(score, dst, nmax, exb, nsum);
        k_message<<<cdiv((long)EE * HIDD, 256), 256, 0, stream>>>(
            P, edge_feats, src, dst, msg_edg_w, msg_edg_b, exb, nsum, agg);
        k_update<<<cdiv((long)NN * HIDD, 256), 256, 0, stream>>>(agg, P, x0, h32, h16);
    }

    k_nodeh<<<cdiv((long)NN * GF, 256), 256, 0, stream>>>(
        node_feats, h32, atom_out_w, atom_out_b, nh);
    k_wsum<<<cdiv(NN, 256), 256, 0, stream>>>(nh, w_sum_w, w_sum_b, wn);
    k_init_hg<<<cdiv(BB * GF, 256), 256, 0, stream>>>(hgs, hgm);
    k_readout<<<cdiv((long)NN * GF, 256), 256, 0, stream>>>(nh, wn, n2g, hgs, hgm);
    k_final<<<cdiv(BB * PREDD, 256), 256, 0, stream>>>(
        hgs, hgm, trans_w, trans_b, (float*)d_out);
}
